// Mamba2DBlock_17068200035059
// MI455X (gfx1250) — matrix operand order repaired and checked by reading, not compiled, not run on hardware
//
#include <hip/hip_runtime.h>
#include <hip/hip_bf16.h>
#include <math.h>

// ---------------------------------------------------------------------------
// Mamba2D block for MI455X (gfx1250, wave32).
//  - All 4 GEMMs use v_wmma_f32_16x16x32_f16 (f16 in, f32 accumulate),
//    4 M-subtiles per wave so each weight (B) fragment feeds 4 WMMAs.
//  - Selective scan: 1 wave per (batch, channel); 4 states/lane in registers,
//    y-reduction via wave32 shfl_xor butterfly; global_prefetch look-ahead.
// ---------------------------------------------------------------------------

#define D_MODEL 64
#define D_STATE 128
#define D_CONV  4
#define D_INNER 128
#define BATCH   4
#define LSEQ    4096                    // H*W
#define NTOK    (BATCH * LSEQ)          // 16384 tokens
#define XZ_COLS (2 * D_INNER)           // 256
#define XDBL_COLS (D_INNER + 2 * D_STATE) // 384
#define MTILES  4                       // M-subtiles per wave (64 rows / wave)

typedef __attribute__((ext_vector_type(16))) _Float16 v16h;
typedef __attribute__((ext_vector_type(8)))  _Float16 v8h;
typedef __attribute__((ext_vector_type(8)))  float    v8f;

// ---------------------------------------------------------------------------
// fp32 -> fp16 weight conversion
// ---------------------------------------------------------------------------
__global__ void cvt_f16_k(const float* __restrict__ src, _Float16* __restrict__ dst, int n) {
    int i = blockIdx.x * 256 + threadIdx.x;
    if (i < n) dst[i] = (_Float16)src[i];
}

// ---------------------------------------------------------------------------
// LayerNorm over 64 channels of NCHW input; emit f16 tokens [NTOK, 64]
// ---------------------------------------------------------------------------
__global__ void ln_k(const float* __restrict__ x, const float* __restrict__ g,
                     const float* __restrict__ bta, _Float16* __restrict__ xn_h) {
    int t = blockIdx.x * 256 + threadIdx.x;          // token id, grid covers NTOK
    int b = t >> 12;                                 // /4096
    int l = t & 4095;
    const float* xp = x + (size_t)b * D_MODEL * LSEQ + l;   // stride LSEQ per channel
    float s = 0.f, s2 = 0.f;
#pragma unroll
    for (int c = 0; c < D_MODEL; ++c) {
        float v = xp[(size_t)c * LSEQ];
        s += v; s2 += v * v;
    }
    float mu  = s * (1.f / D_MODEL);
    float var = s2 * (1.f / D_MODEL) - mu * mu;
    float rs  = rsqrtf(var + 1e-5f);
    _Float16* op = xn_h + (size_t)t * D_MODEL;
#pragma unroll
    for (int c = 0; c < D_MODEL; ++c) {
        float v = (xp[(size_t)c * LSEQ] - mu) * rs * g[c] + bta[c];
        op[c] = (_Float16)v;
    }
}

// ---------------------------------------------------------------------------
// WMMA GEMM:  C[M,N] = A[M,K](f16) * W[N,K]^T(f16) + bias
//   One wave32 per 64x16 output block: 4 accumulators share each B fragment,
//   so one weight load feeds 4 v_wmma_f32_16x16x32_f16.
//   ACT: 0 = none, 1 = softplus.  MIRROR: also emit f16 copy of cols < mcols.
// ---------------------------------------------------------------------------
template <int ACT, int MIRROR>
__global__ void wmma_gemm_k(const _Float16* __restrict__ A, int lda,
                            const _Float16* __restrict__ W, int K,
                            const float* __restrict__ bias,
                            float* __restrict__ C, int ldc,
                            _Float16* __restrict__ Ch, int ldch, int mcols) {
    const int lane = threadIdx.x;       // 0..31 (wave32)
    const int lo   = lane & 15;
    const int hi   = lane >> 4;         // 0|1: which K-half of the fragment
    const int tm   = blockIdx.x * (16 * MTILES);
    const int tn   = (blockIdx.y * blockDim.y + threadIdx.y) * 16;

    // A fragment source per sub-tile: lane lo = row,
    //   elements 0..7  -> K = kk + hi*16 + {0..7}
    //   elements 8..15 -> K = kk + hi*16 + 8 + {0..7}
    const _Float16* ap[MTILES];
#pragma unroll
    for (int r = 0; r < MTILES; ++r)
        ap[r] = A + (size_t)(tm + 16 * r + lo) * lda + hi * 16;
    // B fragment source: lane lo = out column n; 16 contiguous K values.
    const _Float16* bp = W + (size_t)(tn + lo) * K + hi * 16;

    v8f acc[MTILES];
#pragma unroll
    for (int r = 0; r < MTILES; ++r) acc[r] = (v8f){};

    for (int kk = 0; kk < K; kk += 32) {
        v16h b = *(const v16h*)(bp + kk);           // one weight load ...
#pragma unroll
        for (int r = 0; r < MTILES; ++r) {          // ... feeds 4 WMMAs
            v8h a0 = *(const v8h*)(ap[r] + kk);
            v8h a1 = *(const v8h*)(ap[r] + kk + 8);
            v16h a;
#pragma unroll
            for (int i = 0; i < 8; ++i) { a[i] = a0[i]; a[8 + i] = a1[i]; }
            acc[r] = __builtin_amdgcn_wmma_f32_16x16x32_f16(
                /*neg_a=*/false, a, /*neg_b=*/false, b,
                /*c_mod=*/(short)0, acc[r], /*reuse_a=*/false, /*reuse_b=*/false);
        }
    }

    const int   n  = tn + lo;
    const float bv = bias[n];
#pragma unroll
    for (int r = 0; r < MTILES; ++r) {
#pragma unroll
        for (int v = 0; v < 8; ++v) {
            int   m   = tm + 16 * r + v + hi * 8;   // D layout: VGPR v -> row v (+8 hi)
            float val = acc[r][v] + bv;
            if (ACT == 1) val = (val > 20.f) ? val : log1pf(expf(val));  // softplus
            C[(size_t)m * ldc + n] = val;
            if (MIRROR) {
                if (n < mcols) Ch[(size_t)m * ldch + n] = (_Float16)val;
            }
        }
    }
}

// ---------------------------------------------------------------------------
// Depthwise causal conv(4) along L + bias + SiLU; emit f32 (scan) + f16 (GEMM)
// ---------------------------------------------------------------------------
__global__ void conv_silu_k(const float* __restrict__ xz, const float* __restrict__ cw,
                            const float* __restrict__ cb, float* __restrict__ xs,
                            _Float16* __restrict__ xs_h) {
    int i = blockIdx.x * 256 + threadIdx.x;     // NTOK * D_INNER
    int d = i & (D_INNER - 1);
    int t = i >> 7;
    int l = t & (LSEQ - 1);
    float acc = cb[d];
#pragma unroll
    for (int j = 0; j < D_CONV; ++j) {
        int ll = l - (D_CONV - 1) + j;
        if (ll >= 0)
            acc += cw[d * D_CONV + j] * xz[(size_t)(t - (D_CONV - 1) + j) * XZ_COLS + d];
    }
    float v = acc / (1.f + expf(-acc));          // SiLU
    xs[i]   = v;
    xs_h[i] = (_Float16)v;
}

// ---------------------------------------------------------------------------
// Selective scan: one wave per (b, d); lane owns states s = lane + 32*{0..3}.
// h recurrence is a pure per-lane FMA chain; y-reduce via shfl_xor butterfly.
// ---------------------------------------------------------------------------
__global__ void scan_k(const float* __restrict__ xs,     // [NTOK, D_INNER] = u
                       const float* __restrict__ delta,  // [NTOK, D_INNER]
                       const float* __restrict__ xdbl,   // [NTOK, 384]; B at +128, C at +256
                       const float* __restrict__ A_log,
                       const float* __restrict__ D_par,
                       float* __restrict__ ys) {         // [NTOK, D_INNER]
    const int lane = threadIdx.x;                         // 0..31
    const int w    = blockIdx.x * blockDim.y + threadIdx.y; // 0..511
    const int b    = w >> 7;
    const int d    = w & (D_INNER - 1);

    float A[4], h[4];
#pragma unroll
    for (int j = 0; j < 4; ++j) {
        A[j] = -expf(A_log[d * D_STATE + lane + 32 * j]);
        h[j] = 0.f;
    }
    const float Dp = D_par[d];

    const float* bbase = xdbl + D_INNER;                  // B_sel column offset
    for (int l = 0; l < LSEQ; ++l) {
        const int t = b * LSEQ + l;
        const float ut = xs[(size_t)t * D_INNER + d];     // wave-uniform broadcast
        const float dt = delta[(size_t)t * D_INNER + d];
        const float* bp = bbase + (size_t)t * XDBL_COLS;  // B_sel row (128 f32)
        const float* cp = bp + D_STATE;                   // C_sel row
        // look-ahead prefetch: hides HBM latency on the serial chain
        __builtin_prefetch(bp + 4 * XDBL_COLS + lane, 0, 1);
        const float du = dt * ut;
        float y = 0.f;
#pragma unroll
        for (int j = 0; j < 4; ++j) {
            float bt = bp[lane + 32 * j];
            float ct = cp[lane + 32 * j];
            h[j] = h[j] * (1.f - dt * A[j]) + du * bt;    // critical-path FMA
            y += h[j] * ct;                               // off critical path
        }
#pragma unroll
        for (int off = 16; off >= 1; off >>= 1)           // wave32 butterfly
            y += __shfl_xor(y, off, 32);
        if (lane == 0) ys[(size_t)t * D_INNER + d] = y + Dp * ut;
    }
}

// ---------------------------------------------------------------------------
// Gate: yg = ys * silu(z)  -> f16 for final GEMM
// ---------------------------------------------------------------------------
__global__ void gate_k(const float* __restrict__ ys, const float* __restrict__ xz,
                       _Float16* __restrict__ yg_h) {
    int i = blockIdx.x * 256 + threadIdx.x;     // NTOK * D_INNER
    int d = i & (D_INNER - 1);
    int t = i >> 7;
    float z = xz[(size_t)t * XZ_COLS + D_INNER + d];
    float v = ys[i] * (z / (1.f + expf(-z)));
    yg_h[i] = (_Float16)v;
}

// ---------------------------------------------------------------------------
// Residual + NHWC(token)->NCHW transpose
// ---------------------------------------------------------------------------
__global__ void resid_k(const float* __restrict__ x, const float* __restrict__ outp,
                        float* __restrict__ y) {
    int i  = blockIdx.x * 256 + threadIdx.x;    // BATCH*D_MODEL*LSEQ
    int hw = i & (LSEQ - 1);
    int c  = (i >> 12) & (D_MODEL - 1);
    int b  = i >> 18;
    y[i] = x[i] + outp[(size_t)(b * LSEQ + hw) * D_MODEL + c];
}

// ---------------------------------------------------------------------------
extern "C" void kernel_launch(void* const* d_in, const int* in_sizes, int n_in,
                              void* d_out, int out_size, void* d_ws, size_t ws_size,
                              hipStream_t stream) {
    const float* x      = (const float*)d_in[0];
    const float* ln_g   = (const float*)d_in[1];
    const float* ln_b   = (const float*)d_in[2];
    const float* W_in   = (const float*)d_in[3];
    const float* b_in   = (const float*)d_in[4];
    const float* conv_w = (const float*)d_in[5];
    const float* conv_b = (const float*)d_in[6];
    const float* W_x    = (const float*)d_in[7];
    const float* b_x    = (const float*)d_in[8];
    const float* W_dt   = (const float*)d_in[9];
    const float* b_dt   = (const float*)d_in[10];
    const float* W_out  = (const float*)d_in[11];
    const float* b_out  = (const float*)d_in[12];
    const float* A_log  = (const float*)d_in[13];
    const float* D_par  = (const float*)d_in[14];
    float* out = (float*)d_out;

    // ---- workspace carve-up (aligned 256B) ----
    char*  ws = (char*)d_ws;
    size_t o  = 0;
    auto carve = [&](size_t bytes) { size_t r = o; o = (o + bytes + 255) & ~(size_t)255; return r; };
    _Float16* W_in_h  = (_Float16*)(ws + carve((size_t)XZ_COLS * D_MODEL * 2));
    _Float16* W_x_h   = (_Float16*)(ws + carve((size_t)XDBL_COLS * D_INNER * 2));
    _Float16* W_dt_h  = (_Float16*)(ws + carve((size_t)D_INNER * D_INNER * 2));
    _Float16* W_out_h = (_Float16*)(ws + carve((size_t)D_MODEL * D_INNER * 2));
    _Float16* xn_h    = (_Float16*)(ws + carve((size_t)NTOK * D_MODEL * 2));
    float*    xz      = (float*)   (ws + carve((size_t)NTOK * XZ_COLS * 4));
    float*    xs      = (float*)   (ws + carve((size_t)NTOK * D_INNER * 4));
    _Float16* xs_h    = (_Float16*)(ws + carve((size_t)NTOK * D_INNER * 2));
    float*    xdbl    = (float*)   (ws + carve((size_t)NTOK * XDBL_COLS * 4));
    _Float16* din_h   = (_Float16*)(ws + carve((size_t)NTOK * D_INNER * 2));
    float*    delta   = (float*)   (ws + carve((size_t)NTOK * D_INNER * 4));
    float*    ysb     = (float*)   (ws + carve((size_t)NTOK * D_INNER * 4));
    _Float16* yg_h    = (_Float16*)(ws + carve((size_t)NTOK * D_INNER * 2));
    float*    outp    = (float*)   (ws + carve((size_t)NTOK * D_MODEL * 4));

    // ---- weight conversion to f16 ----
    cvt_f16_k<<<(XZ_COLS * D_MODEL + 255) / 256, 256, 0, stream>>>(W_in, W_in_h, XZ_COLS * D_MODEL);
    cvt_f16_k<<<(XDBL_COLS * D_INNER + 255) / 256, 256, 0, stream>>>(W_x, W_x_h, XDBL_COLS * D_INNER);
    cvt_f16_k<<<(D_INNER * D_INNER + 255) / 256, 256, 0, stream>>>(W_dt, W_dt_h, D_INNER * D_INNER);
    cvt_f16_k<<<(D_MODEL * D_INNER + 255) / 256, 256, 0, stream>>>(W_out, W_out_h, D_MODEL * D_INNER);

    // ---- LayerNorm ----
    ln_k<<<NTOK / 256, 256, 0, stream>>>(x, ln_g, ln_b, xn_h);

    // ---- GEMM1: xz = xn @ W_in^T + b_in   [16384x64]x[64x256] ----
    wmma_gemm_k<0, 0><<<dim3(NTOK / 64, XZ_COLS / 64), dim3(32, 4), 0, stream>>>(
        xn_h, D_MODEL, W_in_h, D_MODEL, b_in, xz, XZ_COLS, nullptr, 0, 0);

    // ---- conv + SiLU ----
    conv_silu_k<<<(NTOK * D_INNER) / 256, 256, 0, stream>>>(xz, conv_w, conv_b, xs, xs_h);

    // ---- GEMM2: x_dbl = xs @ W_x^T + b_x  [16384x128]x[128x384]; mirror cols<128 to f16 ----
    wmma_gemm_k<0, 1><<<dim3(NTOK / 64, XDBL_COLS / 64), dim3(32, 4), 0, stream>>>(
        xs_h, D_INNER, W_x_h, D_INNER, b_x, xdbl, XDBL_COLS, din_h, D_INNER, D_INNER);

    // ---- GEMM3: delta = softplus(delta_in @ W_dt^T + b_dt) ----
    wmma_gemm_k<1, 0><<<dim3(NTOK / 64, D_INNER / 64), dim3(32, 4), 0, stream>>>(
        din_h, D_INNER, W_dt_h, D_INNER, b_dt, delta, D_INNER, nullptr, 0, 0);

    // ---- selective scan: 512 waves (b,d), 8 waves/block ----
    scan_k<<<(BATCH * D_INNER) / 8, dim3(32, 8), 0, stream>>>(xs, delta, xdbl, A_log, D_par, ysb);

    // ---- gate ----
    gate_k<<<(NTOK * D_INNER) / 256, 256, 0, stream>>>(ysb, xz, yg_h);

    // ---- GEMM4: outp = yg @ W_out^T + b_out  [16384x128]x[128x64] ----
    wmma_gemm_k<0, 0><<<dim3(NTOK / 64, D_MODEL / 64), dim3(32, 4), 0, stream>>>(
        yg_h, D_INNER, W_out_h, D_INNER, b_out, outp, D_MODEL, nullptr, 0, 0);

    // ---- residual + transpose back to NCHW ----
    resid_k<<<(BATCH * D_MODEL * LSEQ) / 256, 256, 0, stream>>>(x, outp, out);

    (void)in_sizes; (void)n_in; (void)out_size; (void)ws_size;
}
